// GraphRNA_41601053229359
// MI455X (gfx1250) — matrix operand order repaired and checked by reading, not compiled, not run on hardware
//
#include <hip/hip_runtime.h>
#include <hip/hip_bf16.h>

#define FH 128  // feature dim H

typedef __attribute__((ext_vector_type(16))) __bf16 v16bf;
typedef __attribute__((ext_vector_type(8)))  __bf16 v8bf;
typedef __attribute__((ext_vector_type(8)))  float  v8f;

// float -> bf16 (round-to-nearest-even) via bit ops.
__device__ __forceinline__ __bf16 f2bf(float f) {
    unsigned int u = __builtin_bit_cast(unsigned int, f);
    unsigned int r = (u + 0x7FFFu + ((u >> 16) & 1u)) >> 16;
    unsigned short s = (unsigned short)r;
    return __builtin_bit_cast(__bf16, s);
}

// ---------------------------------------------------------------- fills
__global__ void fill_kernel(float* __restrict__ p, float v, long n) {
    long i = (long)blockIdx.x * blockDim.x + threadIdx.x;
    if (i < n) p[i] = v;
}

// x[i,:] = emb[ids[i],:]
__global__ void gather_rows(const float* __restrict__ emb, const int* __restrict__ ids,
                            float* __restrict__ out, int n) {
    long t = (long)blockIdx.x * blockDim.x + threadIdx.x;
    if (t >= (long)n * FH) return;
    int i = (int)(t >> 7), f = (int)(t & (FH - 1));
    out[t] = emb[(size_t)ids[i] * FH + f];
}

// cnt[dst[e]] += 1
__global__ void count_edges(const int* __restrict__ dst, float* __restrict__ cnt, int E) {
    int e = blockIdx.x * blockDim.x + threadIdx.x;
    if (e < E) atomicAdd(&cnt[dst[e]], 1.0f);
}

// agg[dst[e],:] += x[src[e],:]   (one wave32 per edge, float4 per lane)
__global__ void scatter_feat(const float* __restrict__ x, const int* __restrict__ src,
                             const int* __restrict__ dst, float* __restrict__ agg, int E) {
    long t = (long)blockIdx.x * blockDim.x + threadIdx.x;
    int e = (int)(t >> 5);
    int lane = (int)(t & 31);
    if (e >= E) return;
    int s = src[e], d = dst[e];
    float4 v = ((const float4*)(x + (size_t)s * FH))[lane];
    float* o = agg + (size_t)d * FH + lane * 4;
    atomicAdd(o + 0, v.x); atomicAdd(o + 1, v.y);
    atomicAdd(o + 2, v.z); atomicAdd(o + 3, v.w);
}

// rs[i] = 1 / max(cnt[i], 1)
__global__ void inv_count(const float* __restrict__ cnt, float* __restrict__ rs, int n) {
    int i = blockIdx.x * blockDim.x + threadIdx.x;
    if (i < n) rs[i] = 1.0f / fmaxf(cnt[i], 1.0f);
}

// out[dst,:] += rsqrt(deg[src])*rsqrt(deg[dst]) * h[src,:]
__global__ void gcn_scatter(const float* __restrict__ h, const int* __restrict__ src,
                            const int* __restrict__ dst, const float* __restrict__ deg,
                            float* __restrict__ out, int E) {
    long t = (long)blockIdx.x * blockDim.x + threadIdx.x;
    int e = (int)(t >> 5);
    int lane = (int)(t & 31);
    if (e >= E) return;
    int s = src[e], d = dst[e];
    float coef = rsqrtf(deg[s]) * rsqrtf(deg[d]);
    float4 v = ((const float4*)(h + (size_t)s * FH))[lane];
    float* o = out + (size_t)d * FH + lane * 4;
    atomicAdd(o + 0, coef * v.x); atomicAdd(o + 1, coef * v.y);
    atomicAdd(o + 2, coef * v.z); atomicAdd(o + 3, coef * v.w);
}

// xm += h0/deg0 + b0 + h1/deg1 + b1 ; relu
__global__ void combine_relu_m(float* __restrict__ xm,
                               const float* __restrict__ h0, const float* __restrict__ deg0,
                               const float* __restrict__ b0,
                               const float* __restrict__ h1, const float* __restrict__ deg1,
                               const float* __restrict__ b1, int n) {
    long t = (long)blockIdx.x * blockDim.x + threadIdx.x;
    if (t >= (long)n * FH) return;
    int i = (int)(t >> 7), f = (int)(t & (FH - 1));
    float v = xm[t] + h0[t] / deg0[i] + b0[f] + h1[t] / deg1[i] + b1[f];
    xm[t] = fmaxf(v, 0.0f);
}

__global__ void relu_kernel(float* __restrict__ x, long n) {
    long i = (long)blockIdx.x * blockDim.x + threadIdx.x;
    if (i < n) x[i] = fmaxf(x[i], 0.0f);
}

// ---------------------------------------------------------------- bf16 prep
// out[t] = bf16(A[t] * rowscale[row])   (row-major M x 128)
__global__ void convert_rows_bf16(const float* __restrict__ A, const float* __restrict__ rowscale,
                                  __bf16* __restrict__ out, long n) {
    long t = (long)blockIdx.x * blockDim.x + threadIdx.x;
    if (t >= n) return;
    int i = (int)(t >> 7);
    float sc = rowscale ? rowscale[i] : 1.0f;
    out[t] = f2bf(A[t] * sc);
}

// Wt[m][n][k] = bf16(W[m][k][n])  -- transpose each 128x128 matrix to column-major
__global__ void convert_wT_bf16(const float* __restrict__ W, __bf16* __restrict__ out, int nmat) {
    long t = (long)blockIdx.x * blockDim.x + threadIdx.x;
    if (t >= (long)nmat * FH * FH) return;
    int m = (int)(t >> 14);
    int r = (int)(t & 16383);
    int n = r >> 7, k = r & 127;
    out[t] = f2bf(W[((size_t)m << 14) + ((size_t)k << 7) + n]);
}

// -------------------------------------------------- WMMA bf16 GEMM
// out[M x 128] (+)= A_bf16[M x 128] @ W (Wt given transposed bf16, [n][k]) (+ bias)
// One 16x16 output tile per wave; K=128 -> 4x v_wmma_f32_16x16x32_bf16.
// Fragment packing per CDNA5 ISA 7.12.2 (16-bit A 16x32 / B 32x16 / f32 D 16x16):
//   A: lane L -> row L&15; half=(L>>4): v0..3 K=half*8+{0..7}, v4..7 K=16+half*8+{0..7}.
//   B: lane L -> col L&15; v0..3 K=half*8+{0..7}, v4..7 K=16+half*8+{0..7} (as A).
//   D: VGPR r -> row (half*8 + r), col = L&15.
// All per-lane fragment data is K-contiguous in our bf16 layouts -> b128 loads.
#define GEMM_WAVES 4
__global__ void gemm128_wmma(const __bf16* __restrict__ A, const __bf16* __restrict__ Wt,
                             const float* __restrict__ bias,
                             float* __restrict__ out, int M, int accumulate) {
    int lane = threadIdx.x & 31;
    int wave = threadIdx.x >> 5;
    int tileM = blockIdx.x * GEMM_WAVES + wave;
    int rowBase = tileM * 16;
    if (rowBase >= M) return;              // wave-uniform: EXEC stays all-ones for WMMA
    int colBase = blockIdx.y * 16;
    int half = lane >> 4;
    int l15  = lane & 15;
    int row  = rowBase + l15;
    int rowc = row < M ? row : (M - 1);
    int col  = colBase + l15;

    const __bf16* Arow = A  + (size_t)rowc * FH;
    const __bf16* Wcol = Wt + (size_t)col  * FH;

    v8f acc = {};
    #pragma unroll
    for (int c = 0; c < 4; ++c) {
        int K0 = c * 32;
        int kbA = K0 + half * 8;
        v8bf a0 = *(const v8bf*)(Arow + kbA);        // K = kbA .. kbA+7
        v8bf a1 = *(const v8bf*)(Arow + kbA + 16);   // K = kbA+16 .. kbA+23
        v16bf a = __builtin_shufflevector(a0, a1, 0, 1, 2, 3, 4, 5, 6, 7,
                                                  8, 9, 10, 11, 12, 13, 14, 15);
        int kbB = K0 + half * 8;
        v8bf b0 = *(const v8bf*)(Wcol + kbB);        // K = kbB .. kbB+7
        v8bf b1 = *(const v8bf*)(Wcol + kbB + 16);   // K = kbB+16 .. kbB+23
        v16bf b = __builtin_shufflevector(b0, b1, 0, 1, 2, 3, 4, 5, 6, 7,
                                                  8, 9, 10, 11, 12, 13, 14, 15);
        acc = __builtin_amdgcn_wmma_f32_16x16x32_bf16(
            false, a, false, b, (short)0, acc, false, false);
    }

    float bv = bias ? bias[col] : 0.0f;
    int srow0 = rowBase + half * 8;
    #pragma unroll
    for (int r = 0; r < 8; ++r) {
        int orow = srow0 + r;
        if (orow < M) {
            size_t idx = (size_t)orow * FH + col;
            float v = acc[r] + bv;
            out[idx] = accumulate ? (out[idx] + v) : v;
        }
    }
}

// out[j] = dot(xs[ls[j],:], xm[ld[j],:])  -- one wave per pair, prefetch pair p+8
__global__ void classify_kernel(const float* __restrict__ xs, const float* __restrict__ xm,
                                const int* __restrict__ ls, const int* __restrict__ ld,
                                float* __restrict__ out, int L) {
    long t = (long)blockIdx.x * blockDim.x + threadIdx.x;
    int p = (int)(t >> 5);
    int lane = (int)(t & 31);
    if (p >= L) return;
    int a = ls[p], b = ld[p];
    // prefetch the rows this wave needs 8 pairs ahead (random-gather stream)
    int pn = p + 8;
    if (pn < L) {
        __builtin_prefetch(xs + (size_t)ls[pn] * FH + lane * 4, 0, 1);
        __builtin_prefetch(xm + (size_t)ld[pn] * FH + lane * 4, 0, 1);
    }
    float4 u = ((const float4*)(xs + (size_t)a * FH))[lane];
    float4 w = ((const float4*)(xm + (size_t)b * FH))[lane];
    float s = u.x * w.x + u.y * w.y + u.z * w.z + u.w * w.w;
    #pragma unroll
    for (int off = 16; off > 0; off >>= 1) s += __shfl_down(s, off, 32);
    if (lane == 0) out[p] = s;
}

// ---------------------------------------------------------------- host
static inline dim3 g1(long n, int b) { return dim3((unsigned)((n + b - 1) / b)); }

extern "C" void kernel_launch(void* const* d_in, const int* in_sizes, int n_in,
                              void* d_out, int out_size, void* d_ws, size_t ws_size,
                              hipStream_t stream) {
    const float* emb_s   = (const float*)d_in[0];
    const float* emb_m   = (const float*)d_in[1];
    const float* sage_Wl = (const float*)d_in[2];
    const float* sage_bl = (const float*)d_in[3];
    const float* sage_Wr = (const float*)d_in[4];
    const float* gcn_W   = (const float*)d_in[5];
    const float* gcn_b   = (const float*)d_in[6];
    const int*   sid     = (const int*)d_in[7];
    const int*   mid     = (const int*)d_in[8];
    const int*   s2m_src = (const int*)d_in[9];
    const int*   s2m_dst = (const int*)d_in[10];
    const int*   sim_src = (const int*)d_in[11];
    const int*   sim_dst = (const int*)d_in[12];
    const int*   lbl_src = (const int*)d_in[13];
    const int*   lbl_dst = (const int*)d_in[14];

    const int NS = in_sizes[7];
    const int NM = in_sizes[8];
    const int E1 = in_sizes[9];
    const int E2 = in_sizes[11];
    const int L  = in_sizes[13];

    float* ws = (float*)d_ws;
    size_t o = 0;
    float* xsA  = ws + o; o += (size_t)NS * FH;
    float* xmA  = ws + o; o += (size_t)NM * FH;
    float* xsB  = ws + o; o += (size_t)NS * FH;
    float* xmB  = ws + o; o += (size_t)NM * FH;
    float* aggm = ws + o; o += (size_t)NM * FH;
    float* aggs = ws + o; o += (size_t)NS * FH;
    float* h0   = ws + o; o += (size_t)NM * FH;
    float* h1   = ws + o; o += (size_t)NM * FH;
    float* cntm = ws + o; o += NM;
    float* cnts = ws + o; o += NS;
    float* rsm  = ws + o; o += NM;
    float* rss  = ws + o; o += NS;
    float* deg0 = ws + o; o += NM;
    float* deg1 = ws + o; o += NM;
    // bf16 regions (carved in float units; all offsets stay 16B-aligned)
    __bf16* WlT = (__bf16*)(ws + o); o += (size_t)4 * FH * FH / 2;   // 4 mats
    __bf16* WrT = (__bf16*)(ws + o); o += (size_t)4 * FH * FH / 2;
    __bf16* gWT = (__bf16*)(ws + o); o += (size_t)4 * FH * FH / 2;
    __bf16* abf = (__bf16*)(ws + o); o += (size_t)NM * FH / 2;       // A operand staging

    const int B = 256;

    // one-time: transposed bf16 weights
    convert_wT_bf16<<<g1((long)4 * FH * FH, B), B, 0, stream>>>(sage_Wl, WlT, 4);
    convert_wT_bf16<<<g1((long)4 * FH * FH, B), B, 0, stream>>>(sage_Wr, WrT, 4);
    convert_wT_bf16<<<g1((long)4 * FH * FH, B), B, 0, stream>>>(gcn_W,   gWT, 4);

    // x = emb[ids]
    gather_rows<<<g1((long)NS * FH, B), B, 0, stream>>>(emb_s, sid, xsA, NS);
    gather_rows<<<g1((long)NM * FH, B), B, 0, stream>>>(emb_m, mid, xmA, NM);

    float* xs_in = xsA; float* xm_in = xmA;
    float* xs_out = xsB; float* xm_out = xmB;

    for (int l = 0; l < 2; ++l) {
        const __bf16* WlT0 = WlT + (size_t)(l * 2 + 0) * FH * FH;
        const __bf16* WlT1 = WlT + (size_t)(l * 2 + 1) * FH * FH;
        const __bf16* WrT0 = WrT + (size_t)(l * 2 + 0) * FH * FH;
        const __bf16* WrT1 = WrT + (size_t)(l * 2 + 1) * FH * FH;
        const __bf16* gWT0 = gWT + (size_t)(l * 2 + 0) * FH * FH;
        const __bf16* gWT1 = gWT + (size_t)(l * 2 + 1) * FH * FH;
        const float*  bl0  = sage_bl + (size_t)(l * 2 + 0) * FH;
        const float*  bl1  = sage_bl + (size_t)(l * 2 + 1) * FH;
        const float*  gb0  = gcn_b + (size_t)(l * 2 + 0) * FH;
        const float*  gb1  = gcn_b + (size_t)(l * 2 + 1) * FH;

        // init scratch
        fill_kernel<<<g1((long)NM * FH, B), B, 0, stream>>>(aggm, 0.0f, (long)NM * FH);
        fill_kernel<<<g1((long)NS * FH, B), B, 0, stream>>>(aggs, 0.0f, (long)NS * FH);
        fill_kernel<<<g1(NM, B), B, 0, stream>>>(cntm, 0.0f, NM);
        fill_kernel<<<g1(NS, B), B, 0, stream>>>(cnts, 0.0f, NS);
        fill_kernel<<<g1(NM, B), B, 0, stream>>>(deg0, 1.0f, NM);   // +1 self loop
        fill_kernel<<<g1(NM, B), B, 0, stream>>>(deg1, 1.0f, NM);

        // degrees / counts
        count_edges<<<g1(E1, B), B, 0, stream>>>(s2m_dst, cntm, E1);
        count_edges<<<g1(E1, B), B, 0, stream>>>(s2m_src, cnts, E1);
        count_edges<<<g1(E2, B), B, 0, stream>>>(sim_dst, deg0, E2);
        count_edges<<<g1(E2, B), B, 0, stream>>>(sim_src, deg1, E2);

        // SAGE neighbor sums (both directions of srna<->mrna)
        scatter_feat<<<g1((long)E1 * 32, B), B, 0, stream>>>(xs_in, s2m_src, s2m_dst, aggm, E1);
        scatter_feat<<<g1((long)E1 * 32, B), B, 0, stream>>>(xm_in, s2m_dst, s2m_src, aggs, E1);
        inv_count<<<g1(NM, B), B, 0, stream>>>(cntm, rsm, NM);
        inv_count<<<g1(NS, B), B, 0, stream>>>(cnts, rss, NS);

        // dense linears (WMMA) -- mrna
        dim3 gm((NM + 16 * GEMM_WAVES - 1) / (16 * GEMM_WAVES), FH / 16);
        dim3 gs((NS + 16 * GEMM_WAVES - 1) / (16 * GEMM_WAVES), FH / 16);
        dim3 gbk(GEMM_WAVES * 32);
        // mean @ Wl + bl
        convert_rows_bf16<<<g1((long)NM * FH, B), B, 0, stream>>>(aggm, rsm, abf, (long)NM * FH);
        gemm128_wmma<<<gm, gbk, 0, stream>>>(abf, WlT0, bl0, xm_out, NM, 0);
        // x_m used by three GEMMs: convert once
        convert_rows_bf16<<<g1((long)NM * FH, B), B, 0, stream>>>(xm_in, nullptr, abf, (long)NM * FH);
        gemm128_wmma<<<gm, gbk, 0, stream>>>(abf, WrT0, nullptr, xm_out, NM, 1);  // += x@Wr
        gemm128_wmma<<<gm, gbk, 0, stream>>>(abf, gWT0, nullptr, h0, NM, 0);      // gcn dir0
        gemm128_wmma<<<gm, gbk, 0, stream>>>(abf, gWT1, nullptr, h1, NM, 0);      // gcn dir1

        // GCN normalized scatters into the mrna accumulator
        gcn_scatter<<<g1((long)E2 * 32, B), B, 0, stream>>>(h0, sim_src, sim_dst, deg0, xm_out, E2);
        gcn_scatter<<<g1((long)E2 * 32, B), B, 0, stream>>>(h1, sim_dst, sim_src, deg1, xm_out, E2);

        // self-loop terms + gcn biases + relu
        combine_relu_m<<<g1((long)NM * FH, B), B, 0, stream>>>(xm_out, h0, deg0, gb0, h1, deg1, gb1, NM);

        // srna branch (SAGE only) + relu
        convert_rows_bf16<<<g1((long)NS * FH, B), B, 0, stream>>>(aggs, rss, abf, (long)NS * FH);
        gemm128_wmma<<<gs, gbk, 0, stream>>>(abf, WlT1, bl1, xs_out, NS, 0);
        convert_rows_bf16<<<g1((long)NS * FH, B), B, 0, stream>>>(xs_in, nullptr, abf, (long)NS * FH);
        gemm128_wmma<<<gs, gbk, 0, stream>>>(abf, WrT1, nullptr, xs_out, NS, 1);
        relu_kernel<<<g1((long)NS * FH, B), B, 0, stream>>>(xs_out, (long)NS * FH);

        // ping-pong
        float* t;
        t = xs_in; xs_in = xs_out; xs_out = t;
        t = xm_in; xm_in = xm_out; xm_out = t;
    }

    // classifier: gathered dot products
    classify_kernel<<<g1((long)L * 32, B), B, 0, stream>>>(xs_in, xm_in, lbl_src, lbl_dst,
                                                           (float*)d_out, L);
}
